// MHSA_66219805769965
// MI455X (gfx1250) — matrix operand order repaired and checked by reading, not compiled, not run on hardware
//
#include <hip/hip_runtime.h>
#include <hip/hip_bf16.h>

typedef __bf16 bf16;
typedef __attribute__((ext_vector_type(4)))  __bf16 v4bf;
typedef __attribute__((ext_vector_type(8)))  __bf16 v8bf;
typedef __attribute__((ext_vector_type(16))) __bf16 v16bf;
typedef __attribute__((ext_vector_type(8)))  float  v8f;
typedef unsigned int u32x4 __attribute__((ext_vector_type(4)));
typedef int          i32x4 __attribute__((ext_vector_type(4)));
typedef int          i32x8 __attribute__((ext_vector_type(8)));

// Tensor Data Mover availability (device pass only; host pass sees 0)
#define ATHENA_TDM 0
#if defined(__has_builtin)
#if __has_builtin(__builtin_amdgcn_tensor_load_to_lds)
#undef ATHENA_TDM
#define ATHENA_TDM 1
#endif
#endif

static __device__ __forceinline__ v16bf cat16(v8bf lo, v8bf hi) {
  return __builtin_shufflevector(lo, hi, 0,1,2,3,4,5,6,7,8,9,10,11,12,13,14,15);
}
static __device__ __forceinline__ v8f vzero8() {
  v8f z;
#pragma unroll
  for (int i = 0; i < 8; ++i) z[i] = 0.0f;
  return z;
}
static __device__ __forceinline__ v8f wmma_bf16(v16bf a, v16bf b, v8f c) {
  return __builtin_amdgcn_wmma_f32_16x16x32_bf16(false, a, false, b, (short)0, c, false, false);
}

// ---------------------------------------------------------------------------
// fp32 -> bf16 (weights)
// ---------------------------------------------------------------------------
__global__ void k_f2bf(const float4* __restrict__ in, v4bf* __restrict__ out, int n4) {
  int i = blockIdx.x * blockDim.x + threadIdx.x;
  if (i < n4) {
    float4 f = in[i];
    v4bf o;
    o[0] = (bf16)f.x; o[1] = (bf16)f.y; o[2] = (bf16)f.z; o[3] = (bf16)f.w;
    out[i] = o;
  }
}

// ---------------------------------------------------------------------------
// x[b][c][n] fp32  ->  xt[b][n][c] bf16   (32x32 LDS tile transpose)
// ---------------------------------------------------------------------------
__global__ __launch_bounds__(256) void k_xpose(const float* __restrict__ x,
                                               bf16* __restrict__ xt) {
  __shared__ bf16 t[32][40];
  const int b = blockIdx.z, c0 = blockIdx.y * 32, n0 = blockIdx.x * 32;
  const int r = threadIdx.x >> 3, q = (threadIdx.x & 7) * 4;

  const float* src = x + ((size_t)(b * 512 + c0) + r) * 4096 + n0 + q;
  float4 f = *(const float4*)src;
  t[r][q] = (bf16)f.x; t[r][q + 1] = (bf16)f.y; t[r][q + 2] = (bf16)f.z; t[r][q + 3] = (bf16)f.w;
  __syncthreads();

  bf16* dst = xt + ((size_t)(b * 4096 + n0 + r)) * 512 + c0 + q;
  v4bf o;
#pragma unroll
  for (int i = 0; i < 4; ++i) o[i] = t[q + i][r];
  *(v4bf*)dst = o;
}

// ---------------------------------------------------------------------------
// GEMM: Out[b][d][n] = sum_c W[d][c] * Xt[b][n][c]   (+bias fp32 for proj)
// A = W (16B-contiguous frags, direct global), B = Xt tile (64n x 64c) staged
// in LDS via TDM (double buffered, TENSORcnt) or cooperative copy fallback.
// Block: 128(d) x 64(n), K step 64; 256 threads = 8 waves, wave w: d-strip w*16.
// ---------------------------------------------------------------------------
template <bool PROJ>
__global__ __launch_bounds__(256) void k_gemm(const bf16* __restrict__ Wm,
                                              const bf16* __restrict__ Xt,
                                              bf16* __restrict__ OutB,
                                              float* __restrict__ OutF,
                                              const float* __restrict__ bias,
                                              int D) {
  constexpr int CIN = 512, N = 4096, STEPS = CIN / 64;
  const int b  = blockIdx.z;
  const int n0 = blockIdx.x * 64;
  const int d0 = blockIdx.y * 128;
  const bf16* Xb = Xt + ((size_t)b * N + n0) * CIN;  // tile origin (n0, c=0)

  __shared__ bf16 sX[2][64][72];  // [n][c] rows: 128B data + 16B pad = 144B

  const int tid = threadIdx.x, lane = tid & 31, wave = tid >> 5;
  const int m = lane & 15, half = lane >> 4;
  const int dw = wave * 16;
  const bf16* Arow = Wm + (size_t)(d0 + dw + m) * CIN;

  v8f acc[4];
#pragma unroll
  for (int j = 0; j < 4; ++j) acc[j] = vzero8();

#if ATHENA_TDM
  // loop-invariant descriptor pieces
  i32x8 g1c;
  g1c[0] = (int)((1u << 16) |   // data_size = 2B
                 (1u << 20) |   // pad_enable
                 (4u << 22) |   // pad_interval: 32 DW = 128B
                 (3u << 25));   // pad_amount:   4 DW = 16B
  g1c[1] = (int)(512u << 16);   // tensor_dim0 = 512 (lo16)
  g1c[2] = (int)(64u  << 16);   // dim0 hi=0 | tensor_dim1 = 64
  g1c[3] = (int)(64u  << 16);   // dim1 hi=0 | tile_dim0 = 64
  g1c[4] = 64;                  // tile_dim1 = 64, tile_dim2 = 0
  g1c[5] = 512;                 // tensor_dim0_stride (elems)
  g1c[6] = 0; g1c[7] = 0;
  i32x4 gz; gz[0] = gz[1] = gz[2] = gz[3] = 0;
#endif

  auto issue_tile = [&](int s) {
#if ATHENA_TDM
    if (wave == 0) {
      unsigned lds_off = (unsigned)(size_t)&sX[s & 1][0][0];
      unsigned long long ga = (unsigned long long)(size_t)(Xb + s * 64);
      u32x4 g0;
      g0[0] = 1u;                                   // count=1 (valid, no gather)
      g0[1] = lds_off;                              // lds_addr (bytes)
      g0[2] = (unsigned)ga;                         // global_addr[31:0]
      g0[3] = (unsigned)((ga >> 32) & 0x01FFFFFFu) | (2u << 30);  // addr[56:32] | type=2
#if __clang_major__ >= 23
      i32x8 gz8;
#pragma unroll
      for (int i = 0; i < 8; ++i) gz8[i] = 0;
      __builtin_amdgcn_tensor_load_to_lds(g0, g1c, gz, gz, gz8, 0);
#else
      __builtin_amdgcn_tensor_load_to_lds(g0, g1c, gz, gz, 0);
#endif
    }
#else
    const int r = tid >> 2, q = (tid & 3) * 16;     // 64 rows x 64c, 16 elems/thread
    const bf16* src = Xb + (size_t)r * CIN + s * 64 + q;
    v8bf v0 = *(const v8bf*)src, v1 = *(const v8bf*)(src + 8);
    *(v8bf*)&sX[s & 1][r][q] = v0;
    *(v8bf*)&sX[s & 1][r][q + 8] = v1;
#endif
  };

  issue_tile(0);
  for (int s = 0; s < STEPS; ++s) {
    if (s + 1 < STEPS) issue_tile(s + 1);
#if ATHENA_TDM
    if (wave == 0) {
      if (s + 1 < STEPS) {
        __builtin_amdgcn_s_wait_tensorcnt(1);  // tile s done, tile s+1 in flight
      } else {
        __builtin_amdgcn_s_wait_tensorcnt(0);  // last tile done
      }
    }
#endif
    __syncthreads();  // tile s visible to all waves

    const int k0 = s * 64;
    v16bf a0 = cat16(*(const v8bf*)(Arow + k0 + half * 16),
                     *(const v8bf*)(Arow + k0 + half * 16 + 8));
    v16bf a1 = cat16(*(const v8bf*)(Arow + k0 + 32 + half * 16),
                     *(const v8bf*)(Arow + k0 + 32 + half * 16 + 8));
#pragma unroll
    for (int j = 0; j < 4; ++j) {
      const bf16* brow = &sX[s & 1][j * 16 + m][0];
      v16bf b0 = cat16(*(const v8bf*)(brow + half * 16),
                       *(const v8bf*)(brow + half * 16 + 8));
      v16bf b1 = cat16(*(const v8bf*)(brow + 32 + half * 16),
                       *(const v8bf*)(brow + 32 + half * 16 + 8));
      acc[j] = wmma_bf16(a0, b0, acc[j]);
      acc[j] = wmma_bf16(a1, b1, acc[j]);
    }
    __syncthreads();  // done reading buf before it is re-filled
  }

  const int dgb = d0 + dw;
#pragma unroll
  for (int j = 0; j < 4; ++j) {
#pragma unroll
    for (int r = 0; r < 8; ++r) {
      int dd = dgb + r + 8 * half;  // D layout: M (=d) = r + 8*(lane>=16)
      int nn = n0 + j * 16 + m;     // N (=n) = lane&15  -> coalesced stores
      if (PROJ) {
        OutF[((size_t)b * D + dd) * N + nn] = acc[j][r] + bias[dd];
      } else {
        OutB[((size_t)b * D + dd) * N + nn] = (bf16)acc[j][r];
      }
    }
  }
}

// ---------------------------------------------------------------------------
// Row L2-norm reciprocals for q,k rows (qkv[b][d'][n], d' in [0,1024))
// ---------------------------------------------------------------------------
__global__ __launch_bounds__(128) void k_norm(const bf16* __restrict__ qkv,
                                              float* __restrict__ norms) {
  const int row = blockIdx.x;  // 0..16383
  const int b = row >> 10, dp = row & 1023;
  const bf16* p = qkv + ((size_t)b * 1536 + dp) * 4096;
  const int tid = threadIdx.x;

  float s = 0.0f;
  for (int i = tid * 8; i < 4096; i += 128 * 8) {
    v8bf v = *(const v8bf*)(p + i);
#pragma unroll
    for (int j = 0; j < 8; ++j) { float f = (float)v[j]; s += f * f; }
  }
  __shared__ float red[128];
  red[tid] = s;
  __syncthreads();
  for (int off = 64; off > 0; off >>= 1) {
    if (tid < off) red[tid] += red[tid + off];
    __syncthreads();
  }
  if (tid == 0) norms[row] = 1.0f / fmaxf(sqrtf(red[0]), 1e-12f);
}

// ---------------------------------------------------------------------------
// Channel attention per (b,h): S = q k^T (K=4096) -> scale -> softmax -> P v.
// Writes avT[b][n][c'] (c' = h*64+c) so proj GEMM reads contiguous B-frags.
// ---------------------------------------------------------------------------
__global__ __launch_bounds__(256) void k_attn(const bf16* __restrict__ qkv,
                                              const float* __restrict__ norms,
                                              bf16* __restrict__ avT) {
  const int bh = blockIdx.x;
  const int b = bh >> 3, h = bh & 7;
  const bf16* qp = qkv + ((size_t)b * 1536 + h * 64) * 4096;
  const bf16* kp = qkv + ((size_t)b * 1536 + 512 + h * 64) * 4096;
  const bf16* vp = qkv + ((size_t)b * 1536 + 1024 + h * 64) * 4096;

  __shared__ float sS[64][65];   // scores / AV staging
  __shared__ bf16  sP[64][72];   // softmax probs [c][d]
  __shared__ bf16  sVt[64][72];  // v tile transposed [n][d]
  __shared__ float sQI[64], sKI[64];

  const int tid = threadIdx.x, lane = tid & 31, wave = tid >> 5;
  const int m = lane & 15, half = lane >> 4;

  if (tid < 64) {
    sQI[tid] = norms[b * 1024 + h * 64 + tid];
    sKI[tid] = norms[b * 1024 + 512 + h * 64 + tid];
  }

  // ---- S = q k^T : both fragments direct from global (n contiguous)
  const int c0 = (wave & 3) * 16;
  const int d0 = (wave >> 2) * 32;
  v8f acc0 = vzero8(), acc1 = vzero8();
  for (int n0 = 0; n0 < 4096; n0 += 32) {
    const bf16* qa = qp + (size_t)(c0 + m) * 4096 + n0 + half * 16;
    v16bf a = cat16(*(const v8bf*)qa, *(const v8bf*)(qa + 8));
    const bf16* k0p = kp + (size_t)(d0 + m) * 4096 + n0 + half * 16;
    const bf16* k1p = kp + (size_t)(d0 + 16 + m) * 4096 + n0 + half * 16;
    v16bf b0 = cat16(*(const v8bf*)k0p, *(const v8bf*)(k0p + 8));
    v16bf b1 = cat16(*(const v8bf*)k1p, *(const v8bf*)(k1p + 8));
    acc0 = wmma_bf16(a, b0, acc0);
    acc1 = wmma_bf16(a, b1, acc1);
  }
  __syncthreads();  // sQI/sKI ready

#pragma unroll
  for (int r = 0; r < 8; ++r) {
    int c = c0 + r + 8 * half;
    int d = d0 + m;
    sS[c][d]      = acc0[r] * 0.125f * sQI[c] * sKI[d];
    sS[c][d + 16] = acc1[r] * 0.125f * sQI[c] * sKI[d + 16];
  }
  __syncthreads();

  // ---- softmax over d
  if (tid < 64) {
    float mx = -3.4e38f;
#pragma unroll 4
    for (int d = 0; d < 64; ++d) mx = fmaxf(mx, sS[tid][d]);
    float sum = 0.0f;
#pragma unroll 4
    for (int d = 0; d < 64; ++d) { float e = __expf(sS[tid][d] - mx); sum += e; sS[tid][d] = e; }
    float inv = 1.0f / sum;
#pragma unroll 4
    for (int d = 0; d < 64; ++d) sP[tid][d] = (bf16)(sS[tid][d] * inv);
  }
  __syncthreads();

  // ---- AV = P(64x64) * V(64x4096)
  const int cp = (wave & 3) * 16;
  const int ns = (wave >> 2) * 32;
  v16bf aP[2];
#pragma unroll
  for (int kk = 0; kk < 2; ++kk) {
    const bf16* pa = &sP[cp + m][kk * 32 + half * 16];
    aP[kk] = cat16(*(const v8bf*)pa, *(const v8bf*)(pa + 8));
  }

  for (int n0 = 0; n0 < 4096; n0 += 64) {
    {  // stage v tile (64d x 64n) transposed into sVt[n][d]
      int dr = tid >> 2, qn = (tid & 3) * 16;
      const bf16* vs = vp + (size_t)dr * 4096 + n0 + qn;
      v8bf v0 = *(const v8bf*)vs, v1 = *(const v8bf*)(vs + 8);
#pragma unroll
      for (int i = 0; i < 8; ++i) {
        sVt[qn + i][dr]     = v0[i];
        sVt[qn + 8 + i][dr] = v1[i];
      }
    }
    __syncthreads();

    v8f pv0 = vzero8(), pv1 = vzero8();
#pragma unroll
    for (int kk = 0; kk < 2; ++kk) {
      const bf16* pb0 = &sVt[ns + m][kk * 32 + half * 16];
      const bf16* pb1 = &sVt[ns + 16 + m][kk * 32 + half * 16];
      v16bf bb0 = cat16(*(const v8bf*)pb0, *(const v8bf*)(pb0 + 8));
      v16bf bb1 = cat16(*(const v8bf*)pb1, *(const v8bf*)(pb1 + 8));
      pv0 = wmma_bf16(aP[kk], bb0, pv0);
      pv1 = wmma_bf16(aP[kk], bb1, pv1);
    }
    // park AV tile in LDS: sS[c][n_local]
#pragma unroll
    for (int r = 0; r < 8; ++r) {
      int c = cp + r + 8 * half;
      sS[c][ns + m]      = pv0[r];
      sS[c][ns + 16 + m] = pv1[r];
    }
    __syncthreads();

    {  // coalesced writeout: avT[b][n][h*64+c]
      int nl = tid >> 2, qc = (tid & 3) * 16;
      bf16* dst = avT + ((size_t)(b * 4096 + n0 + nl)) * 512 + h * 64 + qc;
      v8bf o0, o1;
#pragma unroll
      for (int i = 0; i < 8; ++i) {
        o0[i] = (bf16)sS[qc + i][nl];
        o1[i] = (bf16)sS[qc + 8 + i][nl];
      }
      *(v8bf*)dst = o0;
      *(v8bf*)(dst + 8) = o1;
    }
    __syncthreads();
  }
}

// ---------------------------------------------------------------------------
extern "C" void kernel_launch(void* const* d_in, const int* in_sizes, int n_in,
                              void* d_out, int out_size, void* d_ws, size_t ws_size,
                              hipStream_t stream) {
  (void)in_sizes; (void)n_in; (void)out_size; (void)ws_size;
  const float* x      = (const float*)d_in[0];  // 16 x 512 x 64 x 64
  const float* w_qkv  = (const float*)d_in[1];  // 1536 x 512
  const float* w_proj = (const float*)d_in[2];  // 512 x 512
  const float* b_proj = (const float*)d_in[3];  // 512

  char* ws = (char*)d_ws;
  bf16*  xt    = (bf16*)(ws);                    //  67,108,864 B  [b][n][c]
  bf16*  qkv   = (bf16*)(ws + 67108864ull);      // 201,326,592 B  [b][d][n]
  bf16*  avt   = (bf16*)(ws + 268435456ull);     //  67,108,864 B  [b][n][c]
  bf16*  wqb   = (bf16*)(ws + 335544320ull);     //   1,572,864 B
  bf16*  wpb   = (bf16*)(ws + 337117184ull);     //     524,288 B
  float* norms = (float*)(ws + 337641472ull);    //      65,536 B

  // transpose+convert x; convert weights
  k_xpose<<<dim3(128, 16, 16), 256, 0, stream>>>(x, xt);
  k_f2bf<<<(786432 / 4 + 255) / 256, 256, 0, stream>>>((const float4*)w_qkv, (v4bf*)wqb, 786432 / 4);
  k_f2bf<<<(262144 / 4 + 255) / 256, 256, 0, stream>>>((const float4*)w_proj, (v4bf*)wpb, 262144 / 4);

  // QKV GEMM (D=1536): qkv[b][d][n] already in (which,head,hd,n) layout
  k_gemm<false><<<dim3(64, 12, 16), 256, 0, stream>>>(wqb, xt, qkv, nullptr, nullptr, 1536);

  // q,k row L2-norm reciprocals
  k_norm<<<16384, 128, 0, stream>>>(qkv, norms);

  // channel attention -> avT[b][n][c]
  k_attn<<<128, 256, 0, stream>>>(qkv, norms, avt);

  // output projection + bias -> d_out (B,C,H,W) fp32
  k_gemm<true><<<dim3(64, 4, 16), 256, 0, stream>>>(wpb, avt, nullptr, (float*)d_out, b_proj, 512);
}
